// AFT_19559281066307
// MI455X (gfx1250) — matrix operand order repaired and checked by reading, not compiled, not run on hardware
//
#include <hip/hip_runtime.h>
#include <hip/hip_bf16.h>
#include <math.h>

#define BB 4
#define LL 4096
#define DM 1024
#define DI 2048
#define ML (BB * LL)            // 16384 rows
#define NSPLIT 16
#define ROWS_PER_SPLIT (LL / NSPLIT)   // 256

// GEMM tiling
#define BM 256                  // block tile M (8 waves x 32)
#define BN 64                   // block tile N
#define KSTEP 32
#define AP 40                   // padded LDS row stride (elements) for A tile
#define BP 40                   // padded LDS row stride (elements) for B tile

typedef __attribute__((ext_vector_type(16))) __bf16        v16bf;
typedef __attribute__((ext_vector_type(8)))  float         v8f;
typedef __attribute__((ext_vector_type(4)))  unsigned int  v4u;

union Frag16 { v16bf bf; v4u q[2]; };

// round-to-nearest-even f32 -> bf16 (bit pattern as ushort)
__device__ __forceinline__ unsigned short f2bf(float f) {
    unsigned int u = __float_as_uint(f);
    u += 0x7FFFu + ((u >> 16) & 1u);
    return (unsigned short)(u >> 16);
}

// CDNA5 async copy: 16 bytes global -> LDS, tracked by ASYNCcnt (no VGPR round trip)
__device__ __forceinline__ void async_g2l_b128(unsigned int lds_off, const void* gaddr) {
    asm volatile("global_load_async_to_lds_b128 %0, %1, off"
                 :: "v"(lds_off), "v"((unsigned long long)(uintptr_t)gaddr)
                 : "memory");
}
__device__ __forceinline__ void wait_asynccnt0() {
    asm volatile("s_wait_asynccnt 0x0" ::: "memory");
}

// ---------------- elementwise convert ----------------
__global__ void k_f32_to_bf16(const float* __restrict__ src,
                              unsigned short* __restrict__ dst, int n) {
    int i = blockIdx.x * blockDim.x + threadIdx.x;
    if (i < n) dst[i] = f2bf(src[i]);
}

// W [K,N] f32 row-major -> Wt [N,K] bf16 row-major
__global__ void k_transpose_bf16(const float* __restrict__ W,
                                 unsigned short* __restrict__ Wt, int K, int N) {
    int i = blockIdx.x * blockDim.x + threadIdx.x;
    if (i >= K * N) return;
    int k = i / N, n = i % N;
    Wt[(size_t)n * K + k] = f2bf(W[i]);
}

// ---------------- WMMA GEMM with async-LDS double buffering ----------------
// D[M,N] = A[M,K] * Bt[N,K]^T + bias
// block = 256 threads (8 waves); block tile 256(M) x 64(N); wave tile 32 x 64.
// mode 0: Q epilogue  x = sigmoid(x)/1024 ; mode 1: plain bias add
__global__ __launch_bounds__(256)
void k_gemm_bf16(const unsigned short* __restrict__ A,    // [M,K] bf16 row-major
                 const unsigned short* __restrict__ Bt,   // [N,K] bf16 row-major
                 const float* __restrict__ bias,          // [N]
                 float* __restrict__ D,                   // [M,N] f32
                 int M, int N, int K, int mode) {
    __shared__ unsigned short Asm[2][BM * AP];   // 2 x 256x32 tile (+pad)
    __shared__ unsigned short Bsm[2][BN * BP];   // 2 x  64x32 tile (+pad)

    const int tid  = threadIdx.x;
    const int lane = tid & 31;
    const int wave = tid >> 5;
    const int Mblk = blockIdx.y * BM;
    const int Nblk = blockIdx.x * BN;

    // stage one K-slab (KSTEP wide) of A and B tiles into LDS buffer `buf`
    auto stage = [&](int buf, int kb) {
        // A tile: 256 rows x 4 x 16B chunks = 1024 chunks, 4 per thread
        #pragma unroll
        for (int j = 0; j < 4; ++j) {
            int c   = tid + j * 256;
            int row = c >> 2, cc = c & 3;
            const unsigned short* g = A + (size_t)(Mblk + row) * K + kb + cc * 8;
            unsigned int l = (unsigned int)(uintptr_t)&Asm[buf][row * AP + cc * 8];
            async_g2l_b128(l, g);
        }
        // B tile: 64 rows x 4 chunks = 256 chunks, 1 per thread
        {
            int row = tid >> 2, cc = tid & 3;
            const unsigned short* g = Bt + (size_t)(Nblk + row) * K + kb + cc * 8;
            unsigned int l = (unsigned int)(uintptr_t)&Bsm[buf][row * BP + cc * 8];
            async_g2l_b128(l, g);
        }
    };

    const int mrow  = lane & 15;     // row/col within 16-wide fragment
    const int khalf = lane >> 4;     // lane-group selector in K

    v8f acc[2][4];
    #pragma unroll
    for (int i = 0; i < 2; ++i)
        #pragma unroll
        for (int t = 0; t < 4; ++t) acc[i][t] = (v8f){};

    stage(0, 0);
    wait_asynccnt0();
    __syncthreads();

    const int nsteps = K / KSTEP;
    for (int s = 0; s < nsteps; ++s) {
        const int cur = s & 1;
        if (s + 1 < nsteps) stage(cur ^ 1, (s + 1) * KSTEP);

        // A fragments from LDS: element i -> k = khalf*8 + (i&7) + (i>>3)*16
        v16bf a0, a1;
        {
            const unsigned short* p0 = &Asm[cur][(wave * 32 + mrow) * AP + khalf * 8];
            const unsigned short* p1 = &Asm[cur][(wave * 32 + 16 + mrow) * AP + khalf * 8];
            Frag16 f;
            f.q[0] = *(const v4u*)p0;  f.q[1] = *(const v4u*)(p0 + 16);
            a0 = f.bf;
            f.q[0] = *(const v4u*)p1;  f.q[1] = *(const v4u*)(p1 + 16);
            a1 = f.bf;
        }
        // B fragments: lane = col, element i -> k = khalf*8 + (i&7) + (i>>3)*16
        #pragma unroll
        for (int t = 0; t < 4; ++t) {
            const unsigned short* bp = &Bsm[cur][(t * 16 + mrow) * BP + khalf * 8];
            Frag16 f;
            f.q[0] = *(const v4u*)bp;  f.q[1] = *(const v4u*)(bp + 16);
            v16bf b = f.bf;
            acc[0][t] = __builtin_amdgcn_wmma_f32_16x16x32_bf16(false, a0, false, b, (short)0, acc[0][t], false, false);
            acc[1][t] = __builtin_amdgcn_wmma_f32_16x16x32_bf16(false, a1, false, b, (short)0, acc[1][t], false, false);
        }

        if (s + 1 < nsteps) {
            wait_asynccnt0();     // my prefetch landed
            __syncthreads();      // everyone's prefetch landed / tile free to reuse
        }
    }

    // C/D layout: VGPR j -> m = khalf*8 + j, n = lane&15
    const int n0   = lane & 15;
    const int mtop = Mblk + wave * 32 + khalf * 8;
    #pragma unroll
    for (int i = 0; i < 2; ++i) {
        #pragma unroll
        for (int t = 0; t < 4; ++t) {
            int n = Nblk + t * 16 + n0;
            float bvv = bias[n];
            #pragma unroll
            for (int j = 0; j < 8; ++j) {
                float x = acc[i][t][j] + bvv;
                if (mode == 0) x = (1.0f / (1.0f + __expf(-x))) * (1.0f / 1024.0f);
                D[(size_t)(mtop + i * 16 + j) * N + n] = x;
            }
        }
    }
}

// ---------------- online column softmax + masked K*V reduction (partials) ----------------
__global__ void k_softmax_partial(const float* __restrict__ Kl, const float* __restrict__ V,
                                  const int* __restrict__ mask,
                                  float* __restrict__ pm, float* __restrict__ ps,
                                  float* __restrict__ pn) {
    int d = blockIdx.x * blockDim.x + threadIdx.x;   // channel in D_INT
    int b = blockIdx.y;
    int z = blockIdx.z;
    int r0 = z * ROWS_PER_SPLIT;
    float m = -INFINITY, s = 0.f, num = 0.f;
    for (int r = r0; r < r0 + ROWS_PER_SPLIT; ++r) {
        size_t idx = ((size_t)(b * LL + r)) * DI + d;
        float kl = Kl[idx];
        float v  = V[idx];
        float mk = (float)mask[b * LL + r];
        float nm = fmaxf(m, kl);
        float cs = __expf(m - nm);
        float e  = __expf(kl - nm);
        s   = s * cs + e;
        num = num * cs + e * v * mk;
        m = nm;
    }
    size_t o = ((size_t)z * BB + b) * DI + d;
    pm[o] = m; ps[o] = s; pn[o] = num;
}

__global__ void k_ctx_combine(const float* __restrict__ pm, const float* __restrict__ ps,
                              const float* __restrict__ pn, float* __restrict__ ctx) {
    int i = blockIdx.x * blockDim.x + threadIdx.x;   // b*DI + d
    if (i >= BB * DI) return;
    float M = -INFINITY, S = 0.f, NU = 0.f;
    for (int z = 0; z < NSPLIT; ++z) {
        size_t o = (size_t)z * BB * DI + i;
        float mz = pm[o], sz = ps[o], nz = pn[o];
        float nm = fmaxf(M, mz);
        float c0 = __expf(M - nm), c1 = __expf(mz - nm);
        S  = S  * c0 + sz * c1;
        NU = NU * c0 + nz * c1;
        M = nm;
    }
    ctx[i] = NU / S;
}

// proj = Q * ctx  (bf16 output for the final GEMM)
__global__ void k_scale_q(const float* __restrict__ Q, const float* __restrict__ ctx,
                          unsigned short* __restrict__ P) {
    size_t i = (size_t)blockIdx.x * blockDim.x + threadIdx.x;
    if (i >= (size_t)ML * DI) return;
    int d  = (int)(i % DI);
    int bl = (int)(i / DI);
    int b  = bl / LL;
    P[i] = f2bf(Q[i] * ctx[b * DI + d]);
}

extern "C" void kernel_launch(void* const* d_in, const int* in_sizes, int n_in,
                              void* d_out, int out_size, void* d_ws, size_t ws_size,
                              hipStream_t stream) {
    const float* queries = (const float*)d_in[0];
    const int*   mask    = (const int*)  d_in[1];
    const float* Wq = (const float*)d_in[2];
    const float* bq = (const float*)d_in[3];
    const float* Wk = (const float*)d_in[4];
    const float* bk = (const float*)d_in[5];
    const float* Wv = (const float*)d_in[6];
    const float* bv = (const float*)d_in[7];
    const float* Wo = (const float*)d_in[8];
    const float* bo = (const float*)d_in[9];
    float* out = (float*)d_out;

    char* ws = (char*)d_ws;
    size_t off = 0;
    auto take = [&](size_t bytes) -> char* {
        char* p = ws + off;
        off += (bytes + 255) & ~(size_t)255;
        return p;
    };
    unsigned short* qbf    = (unsigned short*)take((size_t)ML * DM * 2);   // queries bf16
    unsigned short* wqt    = (unsigned short*)take((size_t)DI * DM * 2);   // Wq^T bf16 [DI,DM]
    unsigned short* wkt    = (unsigned short*)take((size_t)DI * DM * 2);
    unsigned short* wvt    = (unsigned short*)take((size_t)DI * DM * 2);
    unsigned short* wot    = (unsigned short*)take((size_t)DM * DI * 2);   // Wo^T bf16 [DM,DI]
    float*          Qf     = (float*)take((size_t)ML * DI * 4);
    float*          Klf    = (float*)take((size_t)ML * DI * 4);
    float*          Vf     = (float*)take((size_t)ML * DI * 4);
    unsigned short* projbf = (unsigned short*)take((size_t)ML * DI * 2);
    float*          ctxb   = (float*)take((size_t)BB * DI * 4);
    float*          pm     = (float*)take((size_t)NSPLIT * BB * DI * 4);
    float*          ps     = (float*)take((size_t)NSPLIT * BB * DI * 4);
    float*          pn     = (float*)take((size_t)NSPLIT * BB * DI * 4);

    const int T = 256;

    // 1) convert activations and weights (weights pre-transposed for contiguous B frags)
    k_f32_to_bf16<<<(ML * DM + T - 1) / T, T, 0, stream>>>(queries, qbf, ML * DM);
    k_transpose_bf16<<<(DM * DI + T - 1) / T, T, 0, stream>>>(Wq, wqt, DM, DI);
    k_transpose_bf16<<<(DM * DI + T - 1) / T, T, 0, stream>>>(Wk, wkt, DM, DI);
    k_transpose_bf16<<<(DM * DI + T - 1) / T, T, 0, stream>>>(Wv, wvt, DM, DI);
    k_transpose_bf16<<<(DI * DM + T - 1) / T, T, 0, stream>>>(Wo, wot, DI, DM);

    // 2) Q/K/V projections (WMMA bf16 -> f32, async-LDS pipelined)
    dim3 g1(DI / BN, ML / BM);          // (32, 64)
    k_gemm_bf16<<<g1, 256, 0, stream>>>(qbf, wqt, bq, Qf,  ML, DI, DM, 0); // sigmoid/1024
    k_gemm_bf16<<<g1, 256, 0, stream>>>(qbf, wkt, bk, Klf, ML, DI, DM, 1); // raw logits
    k_gemm_bf16<<<g1, 256, 0, stream>>>(qbf, wvt, bv, Vf,  ML, DI, DM, 1);

    // 3) ctx[b,d] = sum_r softmax_r(Kl) * V * mask  (online, split over L)
    dim3 gs(DI / T, BB, NSPLIT);
    k_softmax_partial<<<gs, T, 0, stream>>>(Klf, Vf, mask, pm, ps, pn);
    k_ctx_combine<<<(BB * DI + T - 1) / T, T, 0, stream>>>(pm, ps, pn, ctxb);

    // 4) proj = Q * ctx -> bf16
    k_scale_q<<<(int)(((size_t)ML * DI + T - 1) / T), T, 0, stream>>>(Qf, ctxb, projbf);

    // 5) out = proj @ Wo + bo  (WMMA bf16 -> f32)
    dim3 g2(DM / BN, ML / BM);          // (16, 64)
    k_gemm_bf16<<<g2, 256, 0, stream>>>(projbf, wot, bo, out, ML, DM, DI, 1);
}
